// MS3DDeformAttn_29927332118928
// MI455X (gfx1250) — matrix operand order repaired and checked by reading, not compiled, not run on hardware
//
#include <hip/hip_runtime.h>

// ---------------------------------------------------------------------------
// MS 3D deformable attention for MI455X (gfx1250, wave32, WMMA).
// Pipeline: [bf16 WMMA GEMM] x3 -> softmax -> trilinear gather -> [WMMA GEMM]
// ---------------------------------------------------------------------------

#define D_MODEL   256
#define N_FRAMES  3
#define N_LEVELS  4
#define N_T_HEADS 64
#define D_HEAD    4
#define LQ        2048
#define NBATCH    2
#define S_TOT     21760           // 128*128 + 64*64 + 32*32 + 16*16
#define LEN_IN    (N_FRAMES * S_TOT)

typedef __attribute__((ext_vector_type(16))) __bf16 v16bf;
typedef __attribute__((ext_vector_type(8)))  float  v8f;

union FragBF { uint4 u[2]; v16bf v; };

__device__ __forceinline__ unsigned short f32_to_bf16(float f) {
    unsigned int u = __float_as_uint(f);
    u += 0x7FFFu + ((u >> 16) & 1u);   // round-to-nearest-even
    return (unsigned short)(u >> 16);
}

__device__ __forceinline__ unsigned int pack_bf16x2(float lo, float hi) {
    return (unsigned int)f32_to_bf16(lo) | ((unsigned int)f32_to_bf16(hi) << 16);
}

// ---------------------------------------------------------------------------
// Tiled GEMM: C[M,N] = A[M,K] * B[K,N] + bias[N]   (fp32 in/out, bf16 WMMA)
// Block: 128 threads = 4 waves; block tile 64x64; wave tile 32x32 (2x2 WMMA).
// Requires M % 64 == 0, N % 64 == 0, K % 32 == 0 (true for all uses here).
// ---------------------------------------------------------------------------
__global__ __launch_bounds__(128)
void wmma_gemm_bias(const float* __restrict__ A, const float* __restrict__ B,
                    const float* __restrict__ bias, float* __restrict__ C,
                    int M, int N, int K) {
    __shared__ __align__(16) unsigned short As[64][32];   // [m][k] bf16
    __shared__ __align__(16) unsigned short Bs[64][32];   // [n][k] bf16 (transposed)

    const int tid  = threadIdx.x;
    const int lane = tid & 31;
    const int wave = tid >> 5;
    const int wm   = wave >> 1;        // 0..1 : M sub-tile of the wave
    const int wn   = wave & 1;         // 0..1 : N sub-tile of the wave
    const int half = lane >> 4;        // lane group (0: lanes 0-15, 1: 16-31)
    const int l16  = lane & 15;

    const int m0 = blockIdx.y * 64;
    const int n0 = blockIdx.x * 64;

    // per-thread tile-fill coordinates
    const int ar = tid >> 1;                 // A row 0..63
    const int ac = (tid & 1) * 16;           // A k-base 0 / 16
    const int bn = tid & 63;                 // B n 0..63 (consecutive across lanes)
    const int bk = (tid >> 6) * 16;          // B k-base 0 / 16

    const float* Asrc = A + (size_t)(m0 + ar) * K + ac;
    const float* Bsrc = B + n0 + bn + (size_t)bk * N;

    v8f acc[2][2] = {};

    for (int k0 = 0; k0 < K; k0 += 32) {
        // --- prefetch next K-step tiles (global_prefetch_b8)
        if (k0 + 32 < K) {
            __builtin_prefetch(Asrc + k0 + 32, 0, 1);
            __builtin_prefetch(Bsrc + (size_t)32 * N, 0, 1);
        }

        // --- A tile (64 rows x 32 k): float4 loads, packed b128 LDS stores
        {
            const float4* s4 = reinterpret_cast<const float4*>(Asrc + k0);
            float4 f0 = s4[0], f1 = s4[1], f2 = s4[2], f3 = s4[3];
            uint4 lo, hi;
            lo.x = pack_bf16x2(f0.x, f0.y); lo.y = pack_bf16x2(f0.z, f0.w);
            lo.z = pack_bf16x2(f1.x, f1.y); lo.w = pack_bf16x2(f1.z, f1.w);
            hi.x = pack_bf16x2(f2.x, f2.y); hi.y = pack_bf16x2(f2.z, f2.w);
            hi.z = pack_bf16x2(f3.x, f3.y); hi.w = pack_bf16x2(f3.z, f3.w);
            uint4* dst = reinterpret_cast<uint4*>(&As[ar][ac]);
            dst[0] = lo; dst[1] = hi;
        }
        // --- B tile (32 k x 64 n): coalesced column reads, transposed store
        {
            const float* s = Bsrc + (size_t)k0 * N;
            unsigned int pk[8];
            #pragma unroll
            for (int i = 0; i < 8; ++i) {
                const float a0 = s[(size_t)(2 * i)     * N];
                const float a1 = s[(size_t)(2 * i + 1) * N];
                pk[i] = pack_bf16x2(a0, a1);
            }
            uint4* dst = reinterpret_cast<uint4*>(&Bs[bn][bk]);
            dst[0] = make_uint4(pk[0], pk[1], pk[2], pk[3]);
            dst[1] = make_uint4(pk[4], pk[5], pk[6], pk[7]);
        }
        __syncthreads();

        // --- fragment loads from LDS (layouts per CDNA5 ISA 7.12.2)
        v16bf afrag[2], bfrag[2];
        #pragma unroll
        for (int tm = 0; tm < 2; ++tm) {
            const uint4* rowA =
                reinterpret_cast<const uint4*>(&As[wm * 32 + tm * 16 + l16][0]);
            FragBF fa;
            fa.u[0] = rowA[half];       // K = 8*half  .. +7   (elements 0..7)
            fa.u[1] = rowA[2 + half];   // K = 16+8*half .. +7 (elements 8..15)
            afrag[tm] = fa.v;
        }
        #pragma unroll
        for (int tn = 0; tn < 2; ++tn) {
            const uint4* rowB =
                reinterpret_cast<const uint4*>(&Bs[wn * 32 + tn * 16 + l16][0]);
            FragBF fb;
            fb.u[0] = rowB[half];           // K = 8*half .. +7
            fb.u[1] = rowB[2 + half];       // K = 16+8*half .. +7
            bfrag[tn] = fb.v;
        }

        #pragma unroll
        for (int tm = 0; tm < 2; ++tm)
            #pragma unroll
            for (int tn = 0; tn < 2; ++tn)
                acc[tm][tn] = __builtin_amdgcn_wmma_f32_16x16x32_bf16(
                    false, afrag[tm], false, bfrag[tn],
                    (short)0, acc[tm][tn], false, false);

        __syncthreads();
    }

    // --- epilogue: D layout m = e + 8*half, n = lane&15
    #pragma unroll
    for (int tm = 0; tm < 2; ++tm) {
        #pragma unroll
        for (int tn = 0; tn < 2; ++tn) {
            const int n  = n0 + wn * 32 + tn * 16 + l16;
            const float bv = bias[n];
            #pragma unroll
            for (int e = 0; e < 8; ++e) {
                const int m = m0 + wm * 32 + tm * 16 + half * 8 + e;
                C[(size_t)m * N + n] = acc[tm][tn][e] + bv;
            }
        }
    }
}

// ---------------------------------------------------------------------------
// Row-wise softmax over 16 contiguous logits (L*P) per (n,q,head).
// ---------------------------------------------------------------------------
__global__ __launch_bounds__(256)
void softmax16_kernel(float* __restrict__ a, int rows) {
    const int r = blockIdx.x * blockDim.x + threadIdx.x;
    if (r >= rows) return;
    float* p = a + (size_t)r * 16;
    float mx = p[0];
    #pragma unroll
    for (int i = 1; i < 16; ++i) mx = fmaxf(mx, p[i]);
    float e[16];
    float s = 0.f;
    #pragma unroll
    for (int i = 0; i < 16; ++i) { e[i] = __expf(p[i] - mx); s += e[i]; }
    const float inv = 1.0f / s;
    #pragma unroll
    for (int i = 0; i < 16; ++i) p[i] = e[i] * inv;
}

// ---------------------------------------------------------------------------
// Deformable trilinear sampling. One thread per (n, q, th).
// value: (N, T, S_TOT, Th, Dh) f32.  L2 (192 MB) holds the whole tensor.
// ---------------------------------------------------------------------------
__global__ __launch_bounds__(256)
void sample_kernel(const float* __restrict__ rp,     // (N,Lq,L,3)
                   const float* __restrict__ off,    // (N,Lq,Th,L,P,3) = 3072/row
                   const float* __restrict__ attn,   // (N,Lq,Th,16) post-softmax
                   const float* __restrict__ value,  // (N,T,S,Th,Dh)
                   float* __restrict__ outp) {       // (N,Lq,Th,Dh)
    const int HH[N_LEVELS] = {128, 64, 32, 16};
    const int WW[N_LEVELS] = {128, 64, 32, 16};
    const int SS[N_LEVELS] = {0, 16384, 20480, 21504};

    const int g = blockIdx.x * blockDim.x + threadIdx.x;
    if (g >= NBATCH * LQ * N_T_HEADS) return;

    const int th = g & (N_T_HEADS - 1);
    const int nq = g >> 6;                 // n*Lq + q
    const int n  = nq >> 11;               // / Lq

    const float* qrp  = rp   + (size_t)nq * (N_LEVELS * 3);
    const float* qoff = off  + (size_t)nq * 3072 + th * 48;
    const float* qat  = attn + (size_t)nq * 1024 + th * 16;
    const float* vb   = value + (size_t)n * N_FRAMES * S_TOT * D_MODEL + th * D_HEAD;

    float ax = 0.f, ay = 0.f, az = 0.f, aw = 0.f;

    #pragma unroll
    for (int l = 0; l < N_LEVELS; ++l) {
        const int H = HH[l], W = WW[l], s0 = SS[l];
        const float rx = qrp[l * 3 + 0];
        const float ry = qrp[l * 3 + 1];
        const float rt = qrp[l * 3 + 2];
        #pragma unroll
        for (int p = 0; p < 4; ++p) {
            const float* o = qoff + (l * 4 + p) * 3;
            // loc*dim - 0.5 with normalizer (W,H,T) collapses to r*dim + off - 0.5
            const float x = rx * (float)W        + o[0] - 0.5f;
            const float y = ry * (float)H        + o[1] - 0.5f;
            const float t = rt * (float)N_FRAMES + o[2] - 0.5f;
            const float xf = floorf(x), yf = floorf(y), tf = floorf(t);
            const float fx = x - xf, fy = y - yf, ft = t - tf;
            const int x0 = (int)xf, y0 = (int)yf, t0 = (int)tf;
            const float wgt = qat[l * 4 + p];

            #pragma unroll
            for (int dt = 0; dt < 2; ++dt) {
                const int   ti = t0 + dt;
                const float wt = dt ? ft : (1.0f - ft);
                if (ti < 0 || ti >= N_FRAMES) continue;
                #pragma unroll
                for (int dy = 0; dy < 2; ++dy) {
                    const int   yi = y0 + dy;
                    const float wy = dy ? fy : (1.0f - fy);
                    if (yi < 0 || yi >= H) continue;
                    #pragma unroll
                    for (int dx = 0; dx < 2; ++dx) {
                        const int   xi = x0 + dx;
                        const float wx = dx ? fx : (1.0f - fx);
                        if (xi < 0 || xi >= W) continue;
                        const size_t idx =
                            ((size_t)ti * S_TOT + s0 + (size_t)yi * W + xi) * D_MODEL;
                        const float4 gv =
                            *reinterpret_cast<const float4*>(vb + idx);
                        const float w = wgt * wt * wy * wx;
                        ax += gv.x * w; ay += gv.y * w;
                        az += gv.z * w; aw += gv.w * w;
                    }
                }
            }
        }
    }
    float4 r4; r4.x = ax; r4.y = ay; r4.z = az; r4.w = aw;
    *reinterpret_cast<float4*>(outp + (size_t)g * D_HEAD) = r4;
}

// ---------------------------------------------------------------------------
// Host-side launch
// ---------------------------------------------------------------------------
extern "C" void kernel_launch(void* const* d_in, const int* in_sizes, int n_in,
                              void* d_out, int out_size, void* d_ws, size_t ws_size,
                              hipStream_t stream) {
    const float* query  = (const float*)d_in[0];   // (2,2048,256)
    const float* refpts = (const float*)d_in[1];   // (2,2048,4,3)
    const float* inflat = (const float*)d_in[2];   // (2,65280,256)
    const float* W_off  = (const float*)d_in[5];   // (256,3072)
    const float* b_off  = (const float*)d_in[6];
    const float* W_attn = (const float*)d_in[7];   // (256,1024)
    const float* b_attn = (const float*)d_in[8];
    const float* W_val  = (const float*)d_in[9];   // (256,256)
    const float* b_val  = (const float*)d_in[10];
    const float* W_out  = (const float*)d_in[11];  // (256,256)
    const float* b_out  = (const float*)d_in[12];
    float* out = (float*)d_out;

    // workspace carve-up (f32)
    const size_t VAL_ELEMS  = (size_t)NBATCH * LEN_IN * D_MODEL;   // 33,423,360
    const size_t OFF_ELEMS  = (size_t)NBATCH * LQ * 3072;          // 12,582,912
    const size_t ATTN_ELEMS = (size_t)NBATCH * LQ * 1024;          //  4,194,304
    float* ws_val  = (float*)d_ws;
    float* ws_off  = ws_val  + VAL_ELEMS;
    float* ws_attn = ws_off  + OFF_ELEMS;
    float* ws_outp = ws_attn + ATTN_ELEMS;

    const int M_val = NBATCH * LEN_IN;   // 130560 (% 64 == 0)
    const int M_q   = NBATCH * LQ;       //   4096

    dim3 blk(128);

    // 1) value = input_flatten @ W_val + b_val
    wmma_gemm_bias<<<dim3(D_MODEL / 64, M_val / 64), blk, 0, stream>>>(
        inflat, W_val, b_val, ws_val, M_val, D_MODEL, D_MODEL);

    // 2) off = query @ W_off + b_off
    wmma_gemm_bias<<<dim3(3072 / 64, M_q / 64), blk, 0, stream>>>(
        query, W_off, b_off, ws_off, M_q, 3072, D_MODEL);

    // 3) attn logits = query @ W_attn + b_attn
    wmma_gemm_bias<<<dim3(1024 / 64, M_q / 64), blk, 0, stream>>>(
        query, W_attn, b_attn, ws_attn, M_q, 1024, D_MODEL);

    // 4) softmax over L*P=16
    const int rows = M_q * N_T_HEADS;    // 262144
    softmax16_kernel<<<dim3(rows / 256), dim3(256), 0, stream>>>(ws_attn, rows);

    // 5) deformable trilinear sampling
    sample_kernel<<<dim3(rows / 256), dim3(256), 0, stream>>>(
        refpts, ws_off, ws_attn, ws_val, ws_outp);

    // 6) out = out_pre @ W_out + b_out
    wmma_gemm_bias<<<dim3(D_MODEL / 64, M_q / 64), blk, 0, stream>>>(
        ws_outp, W_out, b_out, out, M_q, D_MODEL, D_MODEL);
}
